// PSCU_3118146257344
// MI455X (gfx1250) — matrix operand order repaired and checked by reading, not compiled, not run on hardware
//
#include <hip/hip_runtime.h>
#include <hip/hip_bf16.h>

typedef __attribute__((ext_vector_type(16))) __bf16 v16bf;
typedef __attribute__((ext_vector_type(8)))  __bf16 v8bf;
typedef __attribute__((ext_vector_type(8)))  float  v8f;

#define BB   8
#define NN   2048
#define UPF  4
#define N4   8192
#define KNN  16

// ---------------------------------------------------------------------------
// Tiled WMMA GEMM:  Y[b] = act(W(OxI) * X[b](I x Nn) + bias) [+ residual]
// block = 128 threads (4 waves), tile 64(M) x 64(N), K step 32, bf16 MACs.
// A staged row-major [row][K], B staged col-major [col][K]; K-stride padded
// to 40 bf16 (80 B): fragments are two aligned ds_load_b128 and lane
// addresses step 20 banks (conflict-free across a 16-lane half-wave).
// Staging uses uniform-branch fast paths (float4 loads, no per-elem guards)
// for interior tiles so global loads clause and pipeline.
// ---------------------------------------------------------------------------
#define TM 64
#define TN 64
#define TK 32
#define TKP 40

union FragU { v16bf v; v8bf h[2]; };

__global__ __launch_bounds__(128)
void gemm_wmma_bf16(const float* __restrict__ W, const float* __restrict__ X,
                    const float* __restrict__ bias, const float* __restrict__ R,
                    float* __restrict__ Y,
                    int O, int I, int Nn,
                    long long sX, long long sY, long long sR,
                    int act, int resMode)
{
    __shared__ __bf16 As[TM * TKP];   // [row][k], padded
    __shared__ __bf16 Bs[TN * TKP];   // [col][k], padded

    const int b     = blockIdx.z;
    const int tileN = blockIdx.x * TN;
    const int tileM = blockIdx.y * TM;
    const float* Xb = X + (long long)b * sX;
    float*       Yb = Y + (long long)b * sY;

    const int t    = threadIdx.x;
    const int wave = t >> 5;
    const int lane = t & 31;
    const int mloc = lane & 15;          // row/col within 16-wide fragment
    const int ahk  = (lane >> 4) * 8;    // A K-group base (0 or 8)
    const int bhk  = (lane >> 4) * 8;    // B K-group base (0 or 8)

    const bool mFull  = (tileM + TM) <= O;       // all 64 rows valid
    const bool nFull  = (tileN + TN) <= Nn;      // all 64 cols valid
    const bool aVec   = ((I & 3) == 0);          // W rows 16B-pattern safe
    const bool bVec   = ((Nn & 3) == 0);         // X rows 16B-pattern safe

    v8f zero = {0.f,0.f,0.f,0.f,0.f,0.f,0.f,0.f};
    v8f acc[4];
#pragma unroll
    for (int i = 0; i < 4; ++i) acc[i] = zero;

    for (int k0 = 0; k0 < I; k0 += TK) {
        const bool kFull = (k0 + TK) <= I;

        // ---- stage A (TM x TK) -> As[row][k] ----
        if (mFull && kFull && aVec) {
            // vectorized: float4 loads, contiguous 4xbf16 stores (ds_store_b64)
            for (int e = t; e < TM * TK / 4; e += 128) {
                int r  = e >> 3;             // TK/4 = 8 vec4 per row
                int c4 = (e & 7) << 2;
                const float4 w4 = *(const float4*)(W + (long long)(tileM + r) * I + k0 + c4);
                __bf16* d = &As[r * TKP + c4];
                d[0] = (__bf16)w4.x; d[1] = (__bf16)w4.y;
                d[2] = (__bf16)w4.z; d[3] = (__bf16)w4.w;
            }
        } else if (mFull && kFull) {
            // interior but odd stride: unguarded scalar loads (still clause-able)
            for (int e = t; e < TM * TK; e += 128) {
                int r = e / TK, c = e % TK;
                As[r * TKP + c] = (__bf16)W[(long long)(tileM + r) * I + (k0 + c)];
            }
        } else {
            // edge tile: guarded, zero-padded
            for (int e = t; e < TM * TK; e += 128) {
                int r = e / TK, c = e % TK;
                int gr = tileM + r, gc = k0 + c;
                float v = (gr < O && gc < I) ? W[(long long)gr * I + gc] : 0.0f;
                As[r * TKP + c] = (__bf16)v;
            }
        }

        // ---- stage B (TK x TN) -> Bs[col][k] (transposed) ----
        if (nFull && kFull && bVec) {
            for (int e = t; e < TK * TN / 4; e += 128) {
                int r  = e >> 4;             // TN/4 = 16 vec4 per row
                int c4 = (e & 15) << 2;
                const float4 x4 = *(const float4*)(Xb + (long long)(k0 + r) * Nn + tileN + c4);
                Bs[(c4 + 0) * TKP + r] = (__bf16)x4.x;
                Bs[(c4 + 1) * TKP + r] = (__bf16)x4.y;
                Bs[(c4 + 2) * TKP + r] = (__bf16)x4.z;
                Bs[(c4 + 3) * TKP + r] = (__bf16)x4.w;
            }
        } else if (nFull && kFull) {
            for (int e = t; e < TK * TN; e += 128) {
                int r = e / TN, c = e % TN;
                Bs[c * TKP + r] = (__bf16)Xb[(long long)(k0 + r) * Nn + (tileN + c)];
            }
        } else {
            for (int e = t; e < TK * TN; e += 128) {
                int r = e / TN, c = e % TN;
                int gr = k0 + r, gc = tileN + c;
                float v = (gr < I && gc < Nn) ? Xb[(long long)gr * Nn + gc] : 0.0f;
                Bs[c * TKP + r] = (__bf16)v;
            }
        }
        __syncthreads();

        // A fragment: two aligned 16B runs -> 2x ds_load_b128
        FragU af;
        int arow = wave * 16 + mloc;
        af.h[0] = *(const v8bf*)(&As[arow * TKP + ahk]);
        af.h[1] = *(const v8bf*)(&As[arow * TKP + ahk + 16]);

#pragma unroll
        for (int nt = 0; nt < 4; ++nt) {
            FragU bf;
            int bcol = nt * 16 + mloc;
            const __bf16* bp = &Bs[bcol * TKP + bhk];
            bf.h[0] = *(const v8bf*)(bp);
            bf.h[1] = *(const v8bf*)(bp + 16);
            acc[nt] = __builtin_amdgcn_wmma_f32_16x16x32_bf16(
                false, af.v, false, bf.v, (short)0, acc[nt], false, false);
        }
        __syncthreads();
    }

    // epilogue: bias + activation + residual, bounds-guarded
    const int mbase = (lane >> 4) * 8;
    const int ncol  = lane & 15;
#pragma unroll
    for (int nt = 0; nt < 4; ++nt) {
        int col = tileN + nt * 16 + ncol;
        if (col >= Nn) continue;
#pragma unroll
        for (int r = 0; r < 8; ++r) {
            int row = tileM + wave * 16 + mbase + r;
            if (row >= O) continue;
            float v = acc[nt][r];
            if (bias) v += bias[row];
            if (act == 1)      v = v > 0.f ? v : 0.f;
            else if (act == 2) v = tanhf(v);
            if (resMode == 1)
                v += R[(long long)b * sR + (long long)row * Nn + col];
            else if (resMode == 2)
                v += R[(long long)b * sR + (long long)row * (Nn >> 2) + (col >> 2)];
            Yb[(long long)row * Nn + col] = v;
        }
    }
}

// ---------------------------------------------------------------------------
// Elementwise / reduction / gather kernels
// ---------------------------------------------------------------------------

__global__ __launch_bounds__(256)
void k_gmax(const float* __restrict__ fea, float* __restrict__ gmax)
{
    int bc = blockIdx.x;   // b*128 + c
    const float* p = fea + (long long)bc * NN;
    float m = -3.4e38f;
    for (int i = threadIdx.x; i < NN; i += 256) m = fmaxf(m, p[i]);
    __shared__ float red[256];
    red[threadIdx.x] = m;
    __syncthreads();
    for (int s = 128; s > 0; s >>= 1) {
        if (threadIdx.x < s) red[threadIdx.x] = fmaxf(red[threadIdx.x], red[threadIdx.x + s]);
        __syncthreads();
    }
    if (threadIdx.x == 0) gmax[bc] = red[0];
}

__global__ __launch_bounds__(256)
void k_feat2(const float* __restrict__ fea, const float* __restrict__ gmax,
             const float* __restrict__ gshape, float* __restrict__ out)
{
    long long i = (long long)blockIdx.x * 256 + threadIdx.x;
    const long long total = (long long)BB * 768 * NN;
    if (i >= total) return;
    int n = (int)(i % NN);
    long long t = i / NN;
    int c = (int)(t % 768);
    int b = (int)(t / 768);
    float v;
    if (c < 128)       v = fea[((long long)b * 128 + c) * NN + n];
    else if (c < 256)  v = gmax[b * 128 + (c - 128)];
    else               v = gshape[(long long)b * 512 + (c - 256)];
    out[i] = v;
}

__global__ __launch_bounds__(256)
void k_knn(const float* __restrict__ pos, int* __restrict__ idx)
{
    __shared__ float sx[NN], sy[NN], sz[NN];
    int b = blockIdx.y;
    const float* pb = pos + (long long)b * 3 * NN;
    for (int i = threadIdx.x; i < NN; i += 256) {
        sx[i] = pb[i]; sy[i] = pb[NN + i]; sz[i] = pb[2 * NN + i];
    }
    __syncthreads();
    int n = blockIdx.x * 256 + threadIdx.x;
    float qx = sx[n], qy = sy[n], qz = sz[n];
    float bd[KNN]; int bi[KNN];
#pragma unroll
    for (int j = 0; j < KNN; ++j) { bd[j] = 3.4e38f; bi[j] = 0; }
    for (int m = 0; m < NN; ++m) {
        float dx = qx - sx[m], dy = qy - sy[m], dz = qz - sz[m];
        float d = dx * dx + dy * dy + dz * dz;
        if (d < bd[KNN - 1]) {
            int j = KNN - 1;
            while (j > 0 && bd[j - 1] > d) { bd[j] = bd[j - 1]; bi[j] = bi[j - 1]; --j; }
            bd[j] = d; bi[j] = m;
        }
    }
    int* op = idx + ((long long)b * NN + n) * KNN;
#pragma unroll
    for (int j = 0; j < KNN; ++j) op[j] = bi[j];
}

// qk_rel[b,c,n,j] = q[b,c,n] - k[b,c,idx[b,n,j]]   (C = 64)
__global__ __launch_bounds__(256)
void k_qkrel(const float* __restrict__ q, const float* __restrict__ kk,
             const int* __restrict__ idx, float* __restrict__ out)
{
    long long i = (long long)blockIdx.x * 256 + threadIdx.x;
    const long long total = (long long)BB * 64 * NN * KNN;
    if (i >= total) return;
    int j = (int)(i & (KNN - 1));
    long long t = i >> 4;
    int n = (int)(t % NN); t /= NN;
    int c = (int)(t % 64);
    int b = (int)(t / 64);
    int m = idx[((long long)b * NN + n) * KNN + j];
    out[i] = q[((long long)b * 64 + c) * NN + n] - kk[((long long)b * 64 + c) * NN + m];
}

// pos_rel[b,c,n,j] = pos[b,c,n] - pos[b,c,idx[b,n,j]]   (C = 3)
__global__ __launch_bounds__(256)
void k_posrel(const float* __restrict__ pos, const int* __restrict__ idx,
              float* __restrict__ out)
{
    long long i = (long long)blockIdx.x * 256 + threadIdx.x;
    const long long total = (long long)BB * 3 * NN * KNN;
    if (i >= total) return;
    int j = (int)(i & (KNN - 1));
    long long t = i >> 4;
    int n = (int)(t % NN); t /= NN;
    int c = (int)(t % 3);
    int b = (int)(t / 3);
    int m = idx[((long long)b * NN + n) * KNN + j];
    out[i] = pos[((long long)b * 3 + c) * NN + n] - pos[((long long)b * 3 + c) * NN + m];
}

__global__ __launch_bounds__(256)
void k_add_inplace(float* __restrict__ a, const float* __restrict__ bsrc, long long total)
{
    long long i = (long long)blockIdx.x * 256 + threadIdx.x;
    if (i < total) a[i] += bsrc[i];
}

// softmax over j (16) then agg[b,c,n] = sum_j p_j * (v[b,c,n] + pe[b,c,n,j])
__global__ __launch_bounds__(256)
void k_softmax_agg(const float* __restrict__ attn, const float* __restrict__ pe,
                   const float* __restrict__ v, float* __restrict__ agg)
{
    long long i = (long long)blockIdx.x * 256 + threadIdx.x;
    const long long total = (long long)BB * 64 * NN;
    if (i >= total) return;
    const float* ap = attn + i * KNN;
    const float* pp = pe + i * KNN;
    float mx = -3.4e38f;
#pragma unroll
    for (int j = 0; j < KNN; ++j) mx = fmaxf(mx, ap[j]);
    float e[KNN]; float s = 0.f;
#pragma unroll
    for (int j = 0; j < KNN; ++j) { e[j] = expf(ap[j] - mx); s += e[j]; }
    float inv = 1.f / s;
    float vv = v[i];
    float a = 0.f;
#pragma unroll
    for (int j = 0; j < KNN; ++j) a += e[j] * inv * (vv + pp[j]);
    agg[i] = a;
}

// merged_in = concat([pos(3), manifold(64), rel(128), shape(512)]) -> 707 ch
__global__ __launch_bounds__(256)
void k_merged_in(const float* __restrict__ pos, const float* __restrict__ man,
                 const float* __restrict__ rel, const float* __restrict__ gshape,
                 float* __restrict__ out)
{
    long long i = (long long)blockIdx.x * 256 + threadIdx.x;
    const long long total = (long long)BB * 707 * NN;
    if (i >= total) return;
    int n = (int)(i % NN);
    long long t = i / NN;
    int c = (int)(t % 707);
    int b = (int)(t / 707);
    float v;
    if (c < 3)        v = pos[((long long)b * 3 + c) * NN + n];
    else if (c < 67)  v = man[((long long)b * 64 + (c - 3)) * NN + n];
    else if (c < 195) v = rel[((long long)b * 128 + (c - 67)) * NN + n];
    else              v = gshape[(long long)b * 512 + (c - 195)];
    out[i] = v;
}

// generic 2-buffer channel concat over cols: out(C0+C1 x cols)
__global__ __launch_bounds__(256)
void k_concat2(const float* __restrict__ a, int ca, const float* __restrict__ bsrc, int cb,
               int cols, float* __restrict__ out)
{
    long long i = (long long)blockIdx.x * 256 + threadIdx.x;
    int ct = ca + cb;
    const long long total = (long long)BB * ct * cols;
    if (i >= total) return;
    int n = (int)(i % cols);
    long long t = i / cols;
    int c = (int)(t % ct);
    int b = (int)(t / ct);
    float v = (c < ca) ? a[((long long)b * ca + c) * cols + n]
                       : bsrc[((long long)b * cb + (c - ca)) * cols + n];
    out[i] = v;
}

// repeat along columns x4: out[b,c,p] = in[b,c,p>>2]
__global__ __launch_bounds__(256)
void k_repeat4(const float* __restrict__ in, int C, float* __restrict__ out)
{
    long long i = (long long)blockIdx.x * 256 + threadIdx.x;
    const long long total = (long long)BB * C * N4;
    if (i >= total) return;
    int p = (int)(i % N4);
    long long t = i / N4;
    int c = (int)(t % C);
    int b = (int)(t / C);
    out[i] = in[((long long)b * C + c) * NN + (p >> 2)];
}

// wps[r*64+c] = ps_w[c*256 + r]   (r = o*4 + k)
__global__ __launch_bounds__(256)
void k_wps(const float* __restrict__ psw, float* __restrict__ wps)
{
    int i = blockIdx.x * 256 + threadIdx.x;
    if (i >= 256 * 64) return;
    int c = i & 63;
    int r = i >> 6;
    wps[i] = psw[c * 256 + r];
}

// dmchild[b,o,n*4+k] = pswtmp[b, o*4+k, n]
__global__ __launch_bounds__(256)
void k_psreshape(const float* __restrict__ src, float* __restrict__ dst)
{
    long long i = (long long)blockIdx.x * 256 + threadIdx.x;
    const long long total = (long long)BB * 64 * N4;
    if (i >= total) return;
    int p = (int)(i % N4);
    long long t = i / N4;
    int o = (int)(t % 64);
    int b = (int)(t / 64);
    int n = p >> 2, k = p & 3;
    dst[i] = src[((long long)b * 256 + (o * 4 + k)) * NN + n];
}

// rot6d per child point -> 3x3 row-major rotation (cols = x,y,z)
__global__ __launch_bounds__(256)
void k_rot6d(const float* __restrict__ rots, float* __restrict__ rotm)
{
    int m = blockIdx.x * 256 + threadIdx.x;
    if (m >= BB * N4) return;
    int b = m >> 13;
    int p = m & (N4 - 1);
    int pp = p >> 2;
    const float* rp = rots + (long long)b * 6 * NN;
    float o0 = rp[pp], o1 = rp[NN + pp], o2 = rp[2 * NN + pp];
    float o3 = rp[3 * NN + pp], o4 = rp[4 * NN + pp], o5 = rp[5 * NN + pp];
    float nx = fmaxf(sqrtf(o0 * o0 + o1 * o1 + o2 * o2), 1e-8f);
    float x0 = o0 / nx, x1 = o1 / nx, x2 = o2 / nx;
    float c0 = x1 * o5 - x2 * o4;
    float c1 = x2 * o3 - x0 * o5;
    float c2 = x0 * o4 - x1 * o3;
    float nz = fmaxf(sqrtf(c0 * c0 + c1 * c1 + c2 * c2), 1e-8f);
    float z0 = c0 / nz, z1 = c1 / nz, z2 = c2 / nz;
    float y0 = z1 * x2 - z2 * x1;
    float y1 = z2 * x0 - z0 * x2;
    float y2 = z0 * x1 - z1 * x0;
    float* R = rotm + (long long)m * 9;
    R[0] = x0; R[1] = y0; R[2] = z0;
    R[3] = x1; R[4] = y1; R[5] = z1;
    R[6] = x2; R[7] = y2; R[8] = z2;
}

// final: pos_emb + coef dot -> dw ; duvw rotated ; child_pos = delta + repeat(pos)
__global__ __launch_bounds__(256)
void k_final(const float* __restrict__ uv, const float* __restrict__ coefs,
             const float* __restrict__ rotm, const float* __restrict__ pos,
             float* __restrict__ out_pos, float* __restrict__ out_dw)
{
    int m = blockIdx.x * 256 + threadIdx.x;
    if (m >= BB * N4) return;
    int b = m >> 13;
    int p = m & (N4 - 1);
    int pp = p >> 2;
    float u = uv[((long long)b * 2 + 0) * N4 + p];
    float v = uv[((long long)b * 2 + 1) * N4 + p];
    float xx[4] = {0.f, u, u * u, u * u * u};
    float yy[4] = {0.f, v, v * v, v * v * v};
    float dw = 0.f;
#pragma unroll
    for (int i = 0; i < 4; ++i)
#pragma unroll
        for (int j = 0; j < 4; ++j)
            dw += coefs[((long long)b * 16 + (i * 4 + j)) * NN + pp] * (xx[i] * yy[j] + 1e-6f);
    const float* R = rotm + (long long)m * 9;
    float d0 = R[0] * u + R[1] * v + R[2] * dw;
    float d1 = R[3] * u + R[4] * v + R[5] * dw;
    float d2 = R[6] * u + R[7] * v + R[8] * dw;
    out_pos[((long long)b * 3 + 0) * N4 + p] = d0 + pos[((long long)b * 3 + 0) * NN + pp];
    out_pos[((long long)b * 3 + 1) * N4 + p] = d1 + pos[((long long)b * 3 + 1) * NN + pp];
    out_pos[((long long)b * 3 + 2) * N4 + p] = d2 + pos[((long long)b * 3 + 2) * NN + pp];
    out_dw[m] = dw;
}

// ---------------------------------------------------------------------------
// Host side
// ---------------------------------------------------------------------------
static inline long long cdiv(long long a, long long b) { return (a + b - 1) / b; }

static void gemm(hipStream_t s, const float* W, const float* X, const float* bias,
                 const float* R, float* Y, int O, int I, int Nn,
                 long long sX, long long sY, long long sR, int act, int resMode)
{
    dim3 g((unsigned)cdiv(Nn, TN), (unsigned)cdiv(O, TM), BB);
    gemm_wmma_bf16<<<g, dim3(128), 0, s>>>(W, X, bias, R, Y, O, I, Nn, sX, sY, sR, act, resMode);
}

extern "C" void kernel_launch(void* const* d_in, const int* in_sizes, int n_in,
                              void* d_out, int out_size, void* d_ws, size_t ws_size,
                              hipStream_t stream)
{
    (void)in_sizes; (void)n_in; (void)out_size; (void)ws_size;
    // ---- inputs (pytree flatten: 4 arrays, then params dict sorted by key) ----
    const float* gshape = (const float*)d_in[0];   // (8,512,1)
    const float* ppos   = (const float*)d_in[1];   // (8,3,2048)
    const float* pfea   = (const float*)d_in[2];   // (8,128,2048)
    const float* pman   = (const float*)d_in[3];   // (8,64,2048)
    const float* ch_w0   = (const float*)d_in[4];
    const float* ch_b0   = (const float*)d_in[5];
    const float* ch_w1   = (const float*)d_in[6];
    const float* ch_b1   = (const float*)d_in[7];
    const float* coef_w0 = (const float*)d_in[8];
    const float* coef_b0 = (const float*)d_in[9];
    const float* coef_w1 = (const float*)d_in[10];
    const float* coef_b1 = (const float*)d_in[11];
    const float* dxy_w0  = (const float*)d_in[12];
    const float* dxy_b0  = (const float*)d_in[13];
    const float* dxy_w1  = (const float*)d_in[14];
    const float* dxy_b1  = (const float*)d_in[15];
    const float* man_w0  = (const float*)d_in[16];
    const float* man_b0  = (const float*)d_in[17];
    const float* man_w1  = (const float*)d_in[18];
    const float* man_b1  = (const float*)d_in[19];
    const float* m2_w0   = (const float*)d_in[20];
    const float* m2_b0   = (const float*)d_in[21];
    const float* m2_w1   = (const float*)d_in[22];
    const float* m2_b1   = (const float*)d_in[23];
    const float* ps_w0   = (const float*)d_in[24];
    const float* ps_b0   = (const float*)d_in[25];
    const float* ps_w1   = (const float*)d_in[26];
    const float* ps_b1   = (const float*)d_in[27];
    const float* psw     = (const float*)d_in[28]; // (64,64,4)
    const float* rot_w0  = (const float*)d_in[29];
    const float* rot_b0  = (const float*)d_in[30];
    const float* rot_w1  = (const float*)d_in[31];
    const float* rot_b1  = (const float*)d_in[32];
    const float* sk_w0   = (const float*)d_in[33];
    const float* sk_b0   = (const float*)d_in[34];
    const float* a1_w    = (const float*)d_in[35];
    const float* a1_b    = (const float*)d_in[36];
    const float* a2_w    = (const float*)d_in[37];
    const float* a2_b    = (const float*)d_in[38];
    const float* end_w   = (const float*)d_in[39];
    const float* end_b   = (const float*)d_in[40];
    const float* k_w     = (const float*)d_in[41];
    const float* k_b     = (const float*)d_in[42];
    const float* p1_w    = (const float*)d_in[43];
    const float* p1_b    = (const float*)d_in[44];
    const float* p2_w    = (const float*)d_in[45];
    const float* p2_b    = (const float*)d_in[46];
    const float* q_w     = (const float*)d_in[47];
    const float* q_b     = (const float*)d_in[48];
    const float* st_w    = (const float*)d_in[49];
    const float* st_b    = (const float*)d_in[50];
    const float* v_w     = (const float*)d_in[51];
    const float* v_b     = (const float*)d_in[52];

    // ---- outputs (concatenated flat) ----
    float* out_pos = (float*)d_out;                          // 8*3*8192
    float* out_fea = out_pos + (long long)BB * 3 * N4;       // 8*128*8192
    float* out_man = out_fea + (long long)BB * 128 * N4;     // 8*64*8192
    float* out_dw  = out_man + (long long)BB * 64 * N4;      // 65536

    // ---- workspace bump allocator (with lifetime-safe aliasing) ----
    size_t off = 0;
    auto alloc = [&](size_t nElem) -> float* {
        float* p = (float*)((char*)d_ws + off);
        off = (off + nElem * sizeof(float) + 255) & ~(size_t)255;
        return p;
    };
    float* feat2   = alloc((size_t)BB * 768 * NN);   // alias: merged_in
    float* mlph    = alloc((size_t)BB * 128 * NN);
    float* xyz     = alloc((size_t)BB * 128 * NN);
    float* trh     = alloc((size_t)BB * 64 * NN);
    float* qb      = alloc((size_t)BB * 64 * NN);
    float* kb      = alloc((size_t)BB * 64 * NN);
    float* vb      = alloc((size_t)BB * 64 * NN);
    int*   idx     = (int*)alloc((size_t)BB * NN * KNN);
    float* qk      = alloc((size_t)BB * 64 * NN * KNN);   // alias: pswtmp/dmchild/dxyh/uv
    float* posrel  = alloc((size_t)BB * 3 * NN * KNN);
    float* peh     = alloc((size_t)BB * 64 * NN * KNN);   // alias: attn
    float* pe      = alloc((size_t)BB * 64 * NN * KNN);
    float* attnh   = alloc((size_t)BB * 256 * NN * KNN);  // alias: merged/manh/chin/chh
    float* agg     = alloc((size_t)BB * 64 * NN);
    float* rel     = alloc((size_t)BB * 128 * NN);
    float* manif   = alloc((size_t)BB * 64 * NN);
    float* roth    = alloc((size_t)BB * 16 * NN);
    float* rots    = alloc((size_t)BB * 6 * NN);
    float* rotm    = alloc((size_t)BB * N4 * 9);
    float* psin    = alloc((size_t)BB * 192 * NN);
    float* psh     = alloc((size_t)BB * 64 * NN);
    float* dmfeat  = alloc((size_t)BB * 64 * NN);
    float* wps     = alloc((size_t)256 * 64);
    float* coefh   = alloc((size_t)BB * 16 * NN);
    float* coefs   = alloc((size_t)BB * 16 * NN);
    float* gmax    = alloc((size_t)BB * 128);
    // aliases (dead-buffer reuse; single stream => program-order safe)
    float* mergedin = feat2;                               // 707*NN <= 768*NN
    float* attn     = peh;                                 // peh dead after pos2
    float* merged   = attnh;                               // attnh dead after attn2
    float* manh     = attnh + (size_t)BB * 643 * NN;
    float* chin     = manh  + (size_t)BB * 512 * NN;
    float* chh      = chin  + (size_t)BB * 128 * N4;
    float* pswtmp   = qk;                                  // qk dead after attn1
    float* dmchild  = pswtmp + (size_t)BB * 256 * NN;
    float* dxyh     = dmchild + (size_t)BB * 64 * N4;
    float* uv       = dxyh + (size_t)BB * 16 * N4;

    const long long sF = (long long)128 * NN;   // parent_fea batch stride
    const long long sM = (long long)64 * NN;    // parent_manifold batch stride
    const long long NG = (long long)NN * KNN;   // grouped column count 32768
    auto blocks = [](long long total) { return (unsigned)cdiv(total, 256); };

    // ===== stage 1: feat_2 -> mlp_2 (+residual) =====
    k_gmax<<<BB * 128, 256, 0, stream>>>(pfea, gmax);
    k_feat2<<<blocks((long long)BB * 768 * NN), 256, 0, stream>>>(pfea, gmax, gshape, feat2);
    gemm(stream, m2_w0, feat2, m2_b0, nullptr, mlph, 128, 768, NN, 768LL * NN, sF, 0, 1, 0);
    gemm(stream, m2_w1, mlph, m2_b1, pfea, xyz, 128, 128, NN, sF, sF, sF, 0, 1);

    // ===== stage 2: transformer =====
    gemm(stream, st_w, xyz, st_b, nullptr, trh, 64, 128, NN, sF, sM, 0, 0, 0);
    k_knn<<<dim3(NN / 256, BB), 256, 0, stream>>>(ppos, idx);
    gemm(stream, q_w, trh, q_b, nullptr, qb, 64, 64, NN, sM, sM, 0, 0, 0);
    gemm(stream, k_w, trh, k_b, nullptr, kb, 64, 64, NN, sM, sM, 0, 0, 0);
    gemm(stream, v_w, trh, v_b, nullptr, vb, 64, 64, NN, sM, sM, 0, 0, 0);
    k_qkrel<<<blocks((long long)BB * 64 * NG), 256, 0, stream>>>(qb, kb, idx, qk);
    k_posrel<<<blocks((long long)BB * 3 * NG), 256, 0, stream>>>(ppos, idx, posrel);
    gemm(stream, p1_w, posrel, p1_b, nullptr, peh, 64, 3, (int)NG, 3 * NG, 64 * NG, 0, 1, 0);
    gemm(stream, p2_w, peh, p2_b, nullptr, pe, 64, 64, (int)NG, 64 * NG, 64 * NG, 0, 0, 0);
    k_add_inplace<<<blocks((long long)BB * 64 * NG), 256, 0, stream>>>(qk, pe, (long long)BB * 64 * NG);
    gemm(stream, a1_w, qk, a1_b, nullptr, attnh, 256, 64, (int)NG, 64 * NG, 256 * NG, 0, 1, 0);
    gemm(stream, a2_w, attnh, a2_b, nullptr, attn, 64, 256, (int)NG, 256 * NG, 64 * NG, 0, 0, 0);
    k_softmax_agg<<<blocks((long long)BB * 64 * NN), 256, 0, stream>>>(attn, pe, vb, agg);
    gemm(stream, end_w, agg, end_b, xyz, rel, 128, 64, NN, sM, sF, sF, 0, 1);

    // ===== stage 3: skip / manifold =====
    k_merged_in<<<blocks((long long)BB * 707 * NN), 256, 0, stream>>>(ppos, pman, rel, gshape, mergedin);
    gemm(stream, sk_w0, mergedin, sk_b0, nullptr, merged, 643, 707, NN, 707LL * NN, 643LL * NN, 0, 0, 0);
    gemm(stream, man_w0, merged, man_b0, nullptr, manh, 512, 643, NN, 643LL * NN, 512LL * NN, 0, 1, 0);
    gemm(stream, man_w1, manh, man_b1, pman, manif, 64, 512, NN, 512LL * NN, sM, sM, 0, 1);
    k_repeat4<<<blocks((long long)BB * 64 * N4), 256, 0, stream>>>(manif, 64, out_man);

    // ===== stage 4: rot path =====
    gemm(stream, rot_w0, manif, rot_b0, nullptr, roth, 16, 64, NN, sM, 16LL * NN, 0, 1, 0);
    gemm(stream, rot_w1, roth, rot_b1, nullptr, rots, 6, 16, NN, 16LL * NN, 6LL * NN, 0, 0, 0);
    k_rot6d<<<blocks(BB * N4), 256, 0, stream>>>(rots, rotm);

    // ===== stage 5: point-shuffle path =====
    k_concat2<<<blocks((long long)BB * 192 * NN), 256, 0, stream>>>(rel, 128, manif, 64, NN, psin);
    gemm(stream, ps_w0, psin, ps_b0, nullptr, psh, 64, 192, NN, 192LL * NN, sM, 0, 1, 0);
    gemm(stream, ps_w1, psh, ps_b1, nullptr, dmfeat, 64, 64, NN, sM, sM, 0, 0, 0);
    k_wps<<<cdiv(256 * 64, 256), 256, 0, stream>>>(psw, wps);
    gemm(stream, wps, dmfeat, nullptr, nullptr, pswtmp, 256, 64, NN, sM, 256LL * NN, 0, 0, 0);
    k_psreshape<<<blocks((long long)BB * 64 * N4), 256, 0, stream>>>(pswtmp, dmchild);
    gemm(stream, dxy_w0, dmchild, dxy_b0, nullptr, dxyh, 16, 64, N4, 64LL * N4, 16LL * N4, 0, 1, 0);
    gemm(stream, dxy_w1, dxyh, dxy_b1, nullptr, uv, 2, 16, N4, 16LL * N4, 2LL * N4, 0, 2, 0);

    // ===== stage 6: child features =====
    k_concat2<<<blocks((long long)BB * 128 * N4), 256, 0, stream>>>(dmchild, 64, out_man, 64, N4, chin);
    gemm(stream, ch_w0, chin, ch_b0, nullptr, chh, 128, 128, N4, 128LL * N4, 128LL * N4, 0, 1, 0);
    gemm(stream, ch_w1, chh, ch_b1, xyz, out_fea, 128, 128, N4, 128LL * N4, 128LL * N4, sF, 0, 2);

    // ===== stage 7: coef + final pose =====
    gemm(stream, coef_w0, manif, coef_b0, nullptr, coefh, 16, 64, NN, sM, 16LL * NN, 0, 1, 0);
    gemm(stream, coef_w1, coefh, coef_b1, nullptr, coefs, 16, 16, NN, 16LL * NN, 16LL * NN, 0, 0, 0);
    k_final<<<blocks(BB * N4), 256, 0, stream>>>(uv, coefs, rotm, ppos, out_pos, out_dw);
}
